// LightGCN_LSTM_14980845928568
// MI455X (gfx1250) — matrix operand order repaired and checked by reading, not compiled, not run on hardware
//
#include <hip/hip_runtime.h>

typedef __attribute__((ext_vector_type(16))) _Float16 v16h;
typedef __attribute__((ext_vector_type(8)))  _Float16 v8h;
typedef __attribute__((ext_vector_type(8)))  float    v8f;

#define ED   128   // embed dim
#define GDIM 256   // 4*H = gate dim = 2*ED
#define HID  64    // hidden dim

__device__ __forceinline__ float sigmoidf(float x) { return 1.0f / (1.0f + __expf(-x)); }

// ---------------------------------------------------------------------------
// Init: out = emb*alpha[0], x_cur = emb, x_next = 0, deg = 0
// ---------------------------------------------------------------------------
__global__ void gcn_init(const float* __restrict__ emb, const float* __restrict__ alpha,
                         float* __restrict__ out, float* __restrict__ xcur,
                         float* __restrict__ xnext, float* __restrict__ deg,
                         long n128, int N) {
    long i = (long)blockIdx.x * blockDim.x + threadIdx.x;
    if (i < n128) {
        float e = emb[i];
        out[i]   = e * alpha[0];
        xcur[i]  = e;
        xnext[i] = 0.0f;
    }
    if (i < N) deg[i] = 0.0f;
}

// deg[col[e]] += 1  (float atomic; L2-resident)
__global__ void gcn_degree(const int* __restrict__ col, float* __restrict__ deg, long E) {
    long e = (long)blockIdx.x * blockDim.x + threadIdx.x;
    if (e >= E) return;
    unsafeAtomicAdd(&deg[col[e]], 1.0f);
}

__global__ void gcn_dinv(const float* __restrict__ deg, float* __restrict__ dinv, int N) {
    int i = blockIdx.x * blockDim.x + threadIdx.x;
    if (i >= N) return;
    float d = deg[i];
    dinv[i] = (d > 0.0f) ? rsqrtf(fmaxf(d, 1.0f)) : 0.0f;
}

// ---------------------------------------------------------------------------
// One propagation layer: xout[col] += xin[row] * (dinv[row]*dinv[col])
// 32 threads per edge, float4 per thread -> 4 HW float atomics each.
// 51 MB node arrays sit in the 192 MB L2, so gather+atomic stay on-chip.
// ---------------------------------------------------------------------------
__global__ void gcn_scatter(const int* __restrict__ row, const int* __restrict__ col,
                            const float* __restrict__ dinv,
                            const float* __restrict__ xin, float* __restrict__ xout,
                            long E) {
    long tid = (long)blockIdx.x * blockDim.x + threadIdx.x;
    long e   = tid >> 5;
    if (e >= E) return;
    int  c4  = (int)(tid & 31) * 4;
    int  r   = row[e];
    int  c   = col[e];
    float nrm = dinv[r] * dinv[c];
    const float4 v = *(const float4*)(xin + (long)r * ED + c4);
    float* dst = xout + (long)c * ED + c4;
    unsafeAtomicAdd(dst + 0, v.x * nrm);
    unsafeAtomicAdd(dst + 1, v.y * nrm);
    unsafeAtomicAdd(dst + 2, v.z * nrm);
    unsafeAtomicAdd(dst + 3, v.w * nrm);
}

// out += alpha[aidx]*xnew ; zero xold (it becomes next layer's scatter target)
__global__ void gcn_accum(float* __restrict__ out, const float* __restrict__ xnew,
                          const float* __restrict__ alpha, int aidx,
                          float* __restrict__ xold, long n128) {
    long i = (long)blockIdx.x * blockDim.x + threadIdx.x;
    if (i >= n128) return;
    out[i] += alpha[aidx] * xnew[i];
    xold[i] = 0.0f;
}

// ---------------------------------------------------------------------------
// Build LSTM input directly in f16 (feeds the WMMA GEMM):
//   X[t, 0:128] = out[src]*out[dst] ; X[t, 128:256] = emb_feat[f]
// ---------------------------------------------------------------------------
__global__ void build_x(const int* __restrict__ eli, const int* __restrict__ feat,
                        const float* __restrict__ out, const float* __restrict__ embf,
                        _Float16* __restrict__ X, int T) {
    int t = blockIdx.x;
    int k = threadIdx.x;                 // 0..255
    int s = eli[t];
    int d = eli[T + t];
    int f = feat[t];
    float v;
    if (k < ED) v = out[(long)s * ED + k] * out[(long)d * ED + k];
    else        v = embf[(long)f * ED + (k - ED)];
    X[(long)t * GDIM + k] = (_Float16)v;
}

// One-time f32 -> f16 conversion of w_ih (256x256)
__global__ void convert_w(const float* __restrict__ W, _Float16* __restrict__ Wh, int n) {
    int i = blockIdx.x * blockDim.x + threadIdx.x;
    if (i < n) Wh[i] = (_Float16)W[i];
}

// ---------------------------------------------------------------------------
// pre = X @ w_ih^T + (b_ih + b_hh)   via v_wmma_f32_16x16x32_f16
// One wave per 16x16 tile; fully unrolled K-loop of 8 WMMAs over K=256.
// Lane layouts (A/B: this code's contiguous k order, not the ISA table; D per ISA 7.12.2):
//   A (16x32 f16): row M = lane%16; half=lane/16: elems e -> K=half*16+e
//                  (16B-aligned -> two v8h loads)
//   B (32x16 f16): N = lane%16; elems e -> K = half*16 + e  (32B-aligned v16h)
//   D (16x16 f32): vgpr j: M = j + half*8, N = lane%16
// ---------------------------------------------------------------------------
__global__ void gemm_wmma(const _Float16* __restrict__ X,  // [T, 256] f16
                          const _Float16* __restrict__ Wh, // [256, 256] f16 (w_ih)
                          const float* __restrict__ bih,
                          const float* __restrict__ bhh,
                          float* __restrict__ P,           // [T, 256] f32
                          int T) {
    const int NT = GDIM / 16;  // 16 tiles along gates
    int wave = (blockIdx.x * blockDim.x + threadIdx.x) >> 5;
    int lane = threadIdx.x & 31;
    int mtiles = T / 16;
    if (wave >= mtiles * NT) return;       // wave-uniform guard (EXEC stays all-1 for WMMA)
    int mt = wave / NT, nt = wave % NT;
    int m0 = mt * 16, n0 = nt * 16;
    int half = lane >> 4;
    int r    = lane & 15;

    v8f acc = {};
    const _Float16* arow = X  + (long)(m0 + r) * GDIM;   // A row for this lane
    const _Float16* brow = Wh + (long)(n0 + r) * GDIM;   // B: col n0+r of W^T = row of W
#pragma unroll
    for (int k0 = 0; k0 < GDIM; k0 += 32) {
        const _Float16* ap = arow + k0 + half * 16;
        v8h a0 = *(const v8h*)(ap);
        v8h a1 = *(const v8h*)(ap + 8);
        v16h a;
#pragma unroll
        for (int i = 0; i < 8; i++) { a[i] = a0[i]; a[8 + i] = a1[i]; }
        v16h b = *(const v16h*)(brow + k0 + half * 16);
        acc = __builtin_amdgcn_wmma_f32_16x16x32_f16(
            /*neg_a=*/false, a, /*neg_b=*/false, b,
            /*c_mod=*/(short)0, acc, /*reuse_a=*/false, /*reuse_b=*/false);
    }
    int n = n0 + r;
    float bias = bih[n] + bhh[n];
#pragma unroll
    for (int j = 0; j < 8; j++) {
        int m = m0 + j + half * 8;
        P[(long)m * GDIM + n] = acc[j] + bias;
    }
}

// ---------------------------------------------------------------------------
// Sequential LSTM scan + fused ln projection. Single block, 256 threads.
// Thread g keeps w_hh[g,:] (64 floats) in VGPRs; h broadcast from LDS.
// ---------------------------------------------------------------------------
__global__ void lstm_scan(const float* __restrict__ pre,   // [T, 256]
                          const float* __restrict__ whh,   // [256, 64]
                          const float* __restrict__ lnw,   // [64]
                          const float* __restrict__ lnb,   // [1]
                          float* __restrict__ y, int T) {
    __shared__ float h_s[HID];
    __shared__ float g_s[GDIM];
    __shared__ float red[HID];
    int g = threadIdx.x;

    float w[HID];
#pragma unroll
    for (int k = 0; k < HID; k += 4) {
        float4 v4 = *(const float4*)(whh + (long)g * HID + k);
        w[k] = v4.x; w[k + 1] = v4.y; w[k + 2] = v4.z; w[k + 3] = v4.w;
    }
    float c = 0.0f;
    float lw = (g < HID) ? lnw[g] : 0.0f;
    if (g < HID) h_s[g] = 0.0f;
    __syncthreads();

    for (int t = 0; t < T; t++) {
        float acc = pre[(long)t * GDIM + g];
#pragma unroll
        for (int k = 0; k < HID; k++) acc += w[k] * h_s[k];
        g_s[g] = acc;
        __syncthreads();                 // matvec done; old h no longer needed
        if (g < HID) {
            float ig = sigmoidf(g_s[g]);
            float fg = sigmoidf(g_s[HID + g]);
            float gg = tanhf(g_s[2 * HID + g]);
            float og = sigmoidf(g_s[3 * HID + g]);
            c = fg * c + ig * gg;
            float h = og * tanhf(c);
            h_s[g] = h;
            red[g] = h * lw;
        }
        __syncthreads();                 // new h / red visible
        if (g == 0) {
            float s = lnb[0];
#pragma unroll
            for (int k = 0; k < HID; k++) s += red[k];
            y[t] = s;
        }
        // red is only rewritten after the next iteration's first barrier,
        // so thread 0's read is race-free.
    }
}

// ---------------------------------------------------------------------------
extern "C" void kernel_launch(void* const* d_in, const int* in_sizes, int n_in,
                              void* d_out, int out_size, void* d_ws, size_t ws_size,
                              hipStream_t stream) {
    const int*   edge_index = (const int*)d_in[0];   // [2, E]
    const int*   feature    = (const int*)d_in[1];   // [T]
    const int*   eli        = (const int*)d_in[2];   // [2, T]
    const float* emb        = (const float*)d_in[3]; // [N, 128]
    const float* embf       = (const float*)d_in[4]; // [1000, 128]
    const float* alpha      = (const float*)d_in[5]; // [4]
    const float* w_ih       = (const float*)d_in[6]; // [256, 256]
    const float* w_hh       = (const float*)d_in[7]; // [256, 64]
    const float* b_ih       = (const float*)d_in[8]; // [256]
    const float* b_hh       = (const float*)d_in[9]; // [256]
    const float* ln_w       = (const float*)d_in[10];// [1, 64]
    const float* ln_b       = (const float*)d_in[11];// [1]

    long E    = (long)in_sizes[0] / 2;
    int  T    = in_sizes[1];
    int  N    = in_sizes[3] / ED;
    long n128 = (long)N * ED;

    char*  ws  = (char*)d_ws;
    size_t off = 0;
    auto alloc = [&](size_t bytes) -> void* {
        void* p = ws + off;
        off = (off + bytes + 255) & ~(size_t)255;
        return p;
    };
    float*     deg  = (float*)alloc((size_t)N * 4);
    float*     dinv = (float*)alloc((size_t)N * 4);
    float*     x0   = (float*)alloc((size_t)n128 * 4);
    float*     x1   = (float*)alloc((size_t)n128 * 4);
    float*     outE = (float*)alloc((size_t)n128 * 4);
    _Float16*  Xh   = (_Float16*)alloc((size_t)T * GDIM * 2);
    _Float16*  Wh   = (_Float16*)alloc((size_t)GDIM * GDIM * 2);
    float*     P    = (float*)alloc((size_t)T * GDIM * 4);

    const int* row = edge_index;
    const int* col = edge_index + E;

    int blk = 256;
    long gInit = (n128 + blk - 1) / blk;
    gcn_init<<<(unsigned)gInit, blk, 0, stream>>>(emb, alpha, outE, x0, x1, deg, n128, N);

    long gDeg = (E + blk - 1) / blk;
    gcn_degree<<<(unsigned)gDeg, blk, 0, stream>>>(col, deg, E);

    gcn_dinv<<<(N + blk - 1) / blk, blk, 0, stream>>>(deg, dinv, N);

    // w_ih -> f16 once (overlaps with the GCN layers on the same stream order)
    convert_w<<<(GDIM * GDIM + blk - 1) / blk, blk, 0, stream>>>(w_ih, Wh, GDIM * GDIM);

    float* xa = x0;   // current x
    float* xb = x1;   // zeroed scatter target
    long gScat = (E * 32 + blk - 1) / blk;
    for (int l = 0; l < 3; l++) {
        gcn_scatter<<<(unsigned)gScat, blk, 0, stream>>>(row, col, dinv, xa, xb, E);
        gcn_accum<<<(unsigned)gInit, blk, 0, stream>>>(outE, xb, alpha, l + 1, xa, n128);
        float* tmp = xa; xa = xb; xb = tmp;
    }

    build_x<<<T, GDIM, 0, stream>>>(eli, feature, outE, embf, Xh, T);

    int waves  = (T / 16) * (GDIM / 16);         // 2048 tiles
    int gGemm  = (waves * 32 + blk - 1) / blk;   // 8 waves per block
    gemm_wmma<<<gGemm, blk, 0, stream>>>(Xh, Wh, b_ih, b_hh, P, T);

    lstm_scan<<<1, GDIM, 0, stream>>>(P, w_hh, ln_w, ln_b, (float*)d_out, T);
}
